// EpisodicMemoryModule_27350351741650
// MI455X (gfx1250) — matrix operand order repaired and checked by reading, not compiled, not run on hardware
//
#include <hip/hip_runtime.h>
#include <hip/hip_bf16.h>
#include <math.h>

#define B_   128
#define T_   512
#define U_   256
#define EMB_ 256
#define F4U  1024   /* 4*U */
#define CAT3 768    /* 3*U */

typedef __attribute__((ext_vector_type(16))) _Float16 v16h;
typedef __attribute__((ext_vector_type(8)))  float    v8f;

// ---------------------------------------------------------------------------
// WMMA helpers (CDNA5 wave32, 16x16x32 f16 -> f32 accumulate)
// ---------------------------------------------------------------------------
__device__ __forceinline__ v8f wmma_f16(v16h a, v16h b, v8f c) {
  return __builtin_amdgcn_wmma_f32_16x16x32_f16(false, a, false, b,
                                                (short)0, c, false, false);
}

// A-fragment (16x32, f16) from an LDS row-major [16][ld] f16 tile.
// Layout: lane<16 holds row=lane, K k0..k0+15;
//         lane>=16 holds row=lane-16, K k0+16..k0+31.
__device__ __forceinline__ v16h afrag_lds_f16(const _Float16* base, int ld, int k0) {
  const int lane = threadIdx.x & 31;
  const int row  = lane & 15;
  const int kb   = k0 + ((lane & 16) ? 16 : 0);
  const _Float16* p = base + row * ld + kb;
  v16h a;
#pragma unroll
  for (int i = 0; i < 8; ++i) a[i] = p[i];
#pragma unroll
  for (int i = 0; i < 8; ++i) a[8 + i] = p[8 + i];
  return a;
}

// B-fragment (32x16, f16) from a transposed weight Wt[N][K] (row-major, f16).
// lane<16: column N=col0+lane, K k0..k0+15; lane>=16: col=lane-16, K k0+16..k0+31.
// 32-byte contiguous vector load (ldk and k0 are multiples of 16).
__device__ __forceinline__ v16h bfrag_g(const _Float16* Wt, int ldk, int col0, int k0) {
  const int lane = threadIdx.x & 31;
  const int col  = col0 + (lane & 15);
  const int k    = k0 + ((lane & 16) ? 16 : 0);
  return *(const v16h*)(Wt + (size_t)col * ldk + k);
}

// ---------------------------------------------------------------------------
// Weight prep: transpose [K][N] f32 -> [N][K] f16
// ---------------------------------------------------------------------------
__global__ void k_transpose_f16(const float* __restrict__ src,
                                _Float16* __restrict__ dst, int K, int N) {
  int i = blockIdx.x * blockDim.x + threadIdx.x;
  if (i < K * N) {
    int k = i / N, n = i - k * N;
    dst[(size_t)n * K + k] = (_Float16)src[i];
  }
}

// ---------------------------------------------------------------------------
// xr = facts@Wr + br ; xh = facts@Wh + bh  (step-invariant; computed once)
// grid = B*T/16 workgroups of 512 threads (16 waves); wave n owns N-tile n.
// A tile staged directly as f16 in LDS: no converts in the GEMM loop.
// ---------------------------------------------------------------------------
__global__ __launch_bounds__(512)
void k_xrxh(const float* __restrict__ facts,
            const _Float16* __restrict__ WrT, const _Float16* __restrict__ WhT,
            const float* __restrict__ br, const float* __restrict__ bh,
            float* __restrict__ xr, float* __restrict__ xh) {
  __shared__ _Float16 Af[16 * U_];                      // 8 KB
  const int m0 = blockIdx.x * 16;
  for (int i = threadIdx.x; i < 16 * U_; i += blockDim.x)
    Af[i] = (_Float16)facts[(size_t)(m0 + (i >> 8)) * U_ + (i & 255)];
  __syncthreads();
  const int wave = threadIdx.x >> 5;
  v8f accR = {}, accH = {};
#pragma unroll
  for (int kk = 0; kk < U_ / 32; ++kk) {
    v16h a = afrag_lds_f16(Af, U_, kk * 32);
    accR = wmma_f16(a, bfrag_g(WrT, U_, wave * 16, kk * 32), accR);
    accH = wmma_f16(a, bfrag_g(WhT, U_, wave * 16, kk * 32), accH);
  }
  const int lane = threadIdx.x & 31;
  const int e  = wave * 16 + (lane & 15);
  const int mh = (lane & 16) ? 8 : 0;
#pragma unroll
  for (int v = 0; v < 8; ++v) {
    size_t o = (size_t)(m0 + mh + v) * U_ + e;
    xr[o] = accR[v] + br[e];
    xh[o] = accH[v] + bh[e];
  }
}

// ---------------------------------------------------------------------------
// Fused attention scores: build z tile in LDS (f16), GEMM1 (K=1024) with WMMA,
// tanh, dot with l2_W (shfl + LDS atomic reduce), write raw scores.
// ---------------------------------------------------------------------------
__global__ __launch_bounds__(512)
void k_scores(const float* __restrict__ facts, const float* __restrict__ q,
              const float* __restrict__ mem,  const _Float16* __restrict__ l1WT,
              const float* __restrict__ l1b,  const float* __restrict__ l2W,
              const float* __restrict__ l2b,  float* __restrict__ scores) {
  __shared__ _Float16 Z[16 * F4U];                      // 32 KB
  __shared__ float part[16];
  const int b  = blockIdx.x >> 5;                       // T/16 == 32 tiles per b
  const int t0 = (blockIdx.x & 31) * 16;
  if (threadIdx.x < 16) part[threadIdx.x] = 0.f;
  for (int i = threadIdx.x; i < 16 * F4U; i += blockDim.x) {
    int row = i >> 10, f = i & 1023;
    int reg = f >> 8,  j = f & 255;
    float fv = facts[((size_t)b * T_ + t0 + row) * U_ + j];
    float qv = q[b * U_ + j];
    float mv = mem[b * U_ + j];
    float z = (reg == 0) ? fv * qv
            : (reg == 1) ? fv * mv
            : (reg == 2) ? fabsf(fv - qv) : fabsf(fv - mv);
    Z[i] = (_Float16)z;
  }
  __syncthreads();
  const int wave = threadIdx.x >> 5, lane = threadIdx.x & 31;
  v8f acc = {};
#pragma unroll 4
  for (int kk = 0; kk < F4U / 32; ++kk)
    acc = wmma_f16(afrag_lds_f16(Z, F4U, kk * 32),
                   bfrag_g(l1WT, F4U, wave * 16, kk * 32), acc);
  const int e  = wave * 16 + (lane & 15);
  const int mh = (lane & 16) ? 8 : 0;
  const float w2 = l2W[e];
#pragma unroll
  for (int v = 0; v < 8; ++v) {
    float s = tanhf(acc[v] + l1b[e]) * w2;
    s += __shfl_xor(s, 1, 32);                 // reduce within each 16-lane half
    s += __shfl_xor(s, 2, 32);
    s += __shfl_xor(s, 4, 32);
    s += __shfl_xor(s, 8, 32);
    if ((lane & 15) == 0) atomicAdd(&part[mh + v], s);   // ds_add_f32
  }
  __syncthreads();
  if (threadIdx.x < 16)
    scores[(size_t)b * T_ + t0 + threadIdx.x] = part[threadIdx.x] + l2b[0];
}

// ---------------------------------------------------------------------------
// Softmax over T=512 per batch row; 256 threads, 2 elements each, in place.
// ---------------------------------------------------------------------------
__global__ __launch_bounds__(256)
void k_softmax(float* __restrict__ sc) {
  __shared__ float red[256];
  const int b = blockIdx.x, tid = threadIdx.x;
  float v0 = sc[(size_t)b * T_ + tid];
  float v1 = sc[(size_t)b * T_ + 256 + tid];
  red[tid] = fmaxf(v0, v1);
  __syncthreads();
  for (int s = 128; s > 0; s >>= 1) {
    if (tid < s) red[tid] = fmaxf(red[tid], red[tid + s]);
    __syncthreads();
  }
  float mx = red[0];
  __syncthreads();
  float e0 = expf(v0 - mx), e1 = expf(v1 - mx);
  red[tid] = e0 + e1;
  __syncthreads();
  for (int s = 128; s > 0; s >>= 1) {
    if (tid < s) red[tid] += red[tid + s];
    __syncthreads();
  }
  float inv = 1.0f / red[0];
  sc[(size_t)b * T_ + tid]       = e0 * inv;
  sc[(size_t)b * T_ + 256 + tid] = e1 * inv;
}

// ---------------------------------------------------------------------------
// Sequential attention-GRU scan. Batch rows are independent: 8 workgroups of
// 16 rows each, h kept resident in LDS (f32 + f16 shadow for A-fragments).
// Ur/Uh B-fragments are loop-invariant per wave: hoisted into registers
// (2 x 8 x v16h = 128 VGPRs) so the critical 512-step chain touches only
// LDS + WMMA + VALU, with xr/xh/att streamed from HBM in parallel.
// ---------------------------------------------------------------------------
__global__ __launch_bounds__(512)
void k_gru(const float* __restrict__ xr, const float* __restrict__ xh,
           const float* __restrict__ att,
           const _Float16* __restrict__ UrT, const _Float16* __restrict__ UhT,
           float* __restrict__ episode) {
  __shared__ float    hF[16 * U_];                      // 16 KB
  __shared__ _Float16 hH[16 * U_];                      // 8 KB
  __shared__ _Float16 rh[16 * U_];                      // 8 KB
  const int b0 = blockIdx.x * 16;
  for (int i = threadIdx.x; i < 16 * U_; i += blockDim.x) {
    hF[i] = 0.f; hH[i] = (_Float16)0.f;
  }
  const int wave = threadIdx.x >> 5, lane = threadIdx.x & 31;
  const int e  = wave * 16 + (lane & 15);
  const int mh = (lane & 16) ? 8 : 0;

  v16h bUr[U_ / 32], bUh[U_ / 32];                      // loop-invariant weights
#pragma unroll
  for (int kk = 0; kk < U_ / 32; ++kk) {
    bUr[kk] = bfrag_g(UrT, U_, wave * 16, kk * 32);
    bUh[kk] = bfrag_g(UhT, U_, wave * 16, kk * 32);
  }
  __syncthreads();

  for (int t = 0; t < T_; ++t) {
    v8f acc = {};
#pragma unroll
    for (int kk = 0; kk < U_ / 32; ++kk)
      acc = wmma_f16(afrag_lds_f16(hH, U_, kk * 32), bUr[kk], acc);
#pragma unroll
    for (int v = 0; v < 8; ++v) {
      const int M = mh + v;
      size_t o = ((size_t)(b0 + M) * T_ + t) * U_ + e;
      float r = 1.0f / (1.0f + expf(-(xr[o] + acc[v])));
      rh[M * U_ + e] = (_Float16)(r * hF[M * U_ + e]);
    }
    __syncthreads();
    v8f acc2 = {};
#pragma unroll
    for (int kk = 0; kk < U_ / 32; ++kk)
      acc2 = wmma_f16(afrag_lds_f16(rh, U_, kk * 32), bUh[kk], acc2);
#pragma unroll
    for (int v = 0; v < 8; ++v) {
      const int M = mh + v;
      size_t o = ((size_t)(b0 + M) * T_ + t) * U_ + e;
      float ht = tanhf(xh[o] + acc2[v]);
      float g  = att[(size_t)(b0 + M) * T_ + t];
      float hn = g * ht + (1.0f - g) * hF[M * U_ + e];
      hF[M * U_ + e] = hn;
      hH[M * U_ + e] = (_Float16)hn;
    }
    __syncthreads();
  }
  for (int i = threadIdx.x; i < 16 * U_; i += blockDim.x)
    episode[(size_t)(b0 + (i >> 8)) * U_ + (i & 255)] = hF[i];
}

// ---------------------------------------------------------------------------
// memory = relu([memory, episode, question] @ mem_W + mem_b)  (in place;
// each workgroup only touches its own 16 batch rows, staged in LDS first).
// ---------------------------------------------------------------------------
__global__ __launch_bounds__(512)
void k_mem_update(float* __restrict__ memv, const float* __restrict__ episode,
                  const float* __restrict__ question,
                  const _Float16* __restrict__ mWT, const float* __restrict__ mb) {
  __shared__ _Float16 A[16 * CAT3];                     // 24 KB
  const int b0 = blockIdx.x * 16;
  for (int i = threadIdx.x; i < 16 * CAT3; i += blockDim.x) {
    int row = i / CAT3, c = i - row * CAT3;
    float v;
    if (c < U_)          v = memv[(size_t)(b0 + row) * U_ + c];
    else if (c < 2 * U_) v = episode[(size_t)(b0 + row) * U_ + (c - U_)];
    else                 v = question[(size_t)(b0 + row) * U_ + (c - 2 * U_)];
    A[i] = (_Float16)v;
  }
  __syncthreads();
  const int wave = threadIdx.x >> 5, lane = threadIdx.x & 31;
  v8f acc = {};
#pragma unroll
  for (int kk = 0; kk < CAT3 / 32; ++kk)
    acc = wmma_f16(afrag_lds_f16(A, CAT3, kk * 32),
                   bfrag_g(mWT, CAT3, wave * 16, kk * 32), acc);
  const int e  = wave * 16 + (lane & 15);
  const int mh = (lane & 16) ? 8 : 0;
#pragma unroll
  for (int v = 0; v < 8; ++v)
    memv[(size_t)(b0 + mh + v) * U_ + e] = fmaxf(acc[v] + mb[e], 0.f);
}

// ---------------------------------------------------------------------------
// out = concat([memory, question], axis=1)  -> (128, 512)
// ---------------------------------------------------------------------------
__global__ void k_final(const float* __restrict__ memv,
                        const float* __restrict__ question,
                        float* __restrict__ out) {
  int i = blockIdx.x * blockDim.x + threadIdx.x;
  if (i < B_ * 2 * U_) {
    int b = i >> 9, j = i & 511;
    out[i] = (j < U_) ? memv[(size_t)b * U_ + j]
                      : question[(size_t)b * U_ + (j - U_)];
  }
}

// ---------------------------------------------------------------------------
extern "C" void kernel_launch(void* const* d_in, const int* in_sizes, int n_in,
                              void* d_out, int out_size, void* d_ws, size_t ws_size,
                              hipStream_t stream) {
  const float* facts    = (const float*)d_in[0];
  const float* question = (const float*)d_in[1];
  const float* l1_W     = (const float*)d_in[2];
  const float* l1_b     = (const float*)d_in[3];
  const float* l2_W     = (const float*)d_in[4];
  const float* l2_b     = (const float*)d_in[5];
  const float* Wr       = (const float*)d_in[6];
  const float* Ur       = (const float*)d_in[7];
  const float* br       = (const float*)d_in[8];
  const float* Wh       = (const float*)d_in[9];
  const float* Uh       = (const float*)d_in[10];
  const float* bh       = (const float*)d_in[11];
  const float* mem_W    = (const float*)d_in[12];
  const float* mem_b    = (const float*)d_in[13];
  float* out = (float*)d_out;
  (void)in_sizes; (void)n_in; (void)out_size; (void)ws_size;

  char* ws = (char*)d_ws;
  size_t off = 0;
  auto carve = [&](size_t bytes) -> void* {
    void* p = ws + off;
    off = (off + bytes + 255) & ~(size_t)255;
    return p;
  };
  float*    att     = (float*)carve((size_t)B_ * T_ * 4);
  float*    xr      = (float*)carve((size_t)B_ * T_ * U_ * 4);
  float*    xh      = (float*)carve((size_t)B_ * T_ * U_ * 4);
  float*    episode = (float*)carve((size_t)B_ * U_ * 4);
  float*    memory  = (float*)carve((size_t)B_ * U_ * 4);
  _Float16* l1WT    = (_Float16*)carve((size_t)F4U * EMB_ * 2);
  _Float16* WrT     = (_Float16*)carve((size_t)U_ * U_ * 2);
  _Float16* WhT     = (_Float16*)carve((size_t)U_ * U_ * 2);
  _Float16* UrT     = (_Float16*)carve((size_t)U_ * U_ * 2);
  _Float16* UhT     = (_Float16*)carve((size_t)U_ * U_ * 2);
  _Float16* mWT     = (_Float16*)carve((size_t)CAT3 * U_ * 2);

  const int tpb = 256;
  k_transpose_f16<<<(F4U * EMB_ + tpb - 1) / tpb, tpb, 0, stream>>>(l1_W, l1WT, F4U, EMB_);
  k_transpose_f16<<<(U_ * U_ + tpb - 1) / tpb, tpb, 0, stream>>>(Wr, WrT, U_, U_);
  k_transpose_f16<<<(U_ * U_ + tpb - 1) / tpb, tpb, 0, stream>>>(Wh, WhT, U_, U_);
  k_transpose_f16<<<(U_ * U_ + tpb - 1) / tpb, tpb, 0, stream>>>(Ur, UrT, U_, U_);
  k_transpose_f16<<<(U_ * U_ + tpb - 1) / tpb, tpb, 0, stream>>>(Uh, UhT, U_, U_);
  k_transpose_f16<<<(CAT3 * U_ + tpb - 1) / tpb, tpb, 0, stream>>>(mem_W, mWT, CAT3, U_);

  // Step-invariant GRU input projections (computed once, reused 3x).
  k_xrxh<<<(B_ * T_) / 16, 512, 0, stream>>>(facts, WrT, WhT, br, bh, xr, xh);

  // memory <- question
  hipMemcpyAsync(memory, question, (size_t)B_ * U_ * 4,
                 hipMemcpyDeviceToDevice, stream);

  for (int step = 0; step < 3; ++step) {
    k_scores<<<(B_ * T_) / 16, 512, 0, stream>>>(facts, question, memory,
                                                 l1WT, l1_b, l2_W, l2_b, att);
    k_softmax<<<B_, 256, 0, stream>>>(att);
    k_gru<<<B_ / 16, 512, 0, stream>>>(xr, xh, att, UrT, UhT, episode);
    k_mem_update<<<B_ / 16, 512, 0, stream>>>(memory, episode, question, mWT, mem_b);
  }
  k_final<<<(B_ * 2 * U_ + 255) / 256, 256, 0, stream>>>(memory, question, out);
}
